// RNN_61186104099677
// MI455X (gfx1250) — matrix operand order repaired and checked by reading, not compiled, not run on hardware
//
#include <hip/hip_runtime.h>
#include <stdint.h>

// ---------------------------------------------------------------------------
// Problem dims (fixed by the reference)
// ---------------------------------------------------------------------------
#define EMBED   512
#define HIDDEN  1024
#define SEQ     512
#define BATCH   64
#define MROWS   (BATCH * SEQ)   // 32768 rows of the wx GEMM
#define NCLS    4
#define RNN_WGS 64              // one WG per 16-column slice of HIDDEN
#define ULDS_STRIDE 1032        // 1024 + 8 pad -> 16B aligned rows, bank spread

typedef __attribute__((ext_vector_type(16))) __bf16 v16bf;
typedef __attribute__((ext_vector_type(8)))  __bf16 v8bf;
typedef __attribute__((ext_vector_type(8)))  float  v8f;

// ---------------------------------------------------------------------------
// Fragment builders: native packed conversions + register-block concat.
// A-fragment (16x32 bf16): per lane, two contiguous runs of 8 K-elements.
// B-fragment (32x16 bf16): per lane, two contiguous runs of 8 K-elements
// (same K order as the A-fragment).
// ---------------------------------------------------------------------------
__device__ __forceinline__ v16bf frag_from_f32(const float* p, const float* q) {
    v8f lo = *(const v8f*)p;            // 32B contiguous, 32B-aligned
    v8f hi = *(const v8f*)q;
    v8bf blo = __builtin_convertvector(lo, v8bf);   // v_cvt_pk_bf16_f32 x4
    v8bf bhi = __builtin_convertvector(hi, v8bf);
    return __builtin_shufflevector(blo, bhi,
        0, 1, 2, 3, 4, 5, 6, 7, 8, 9, 10, 11, 12, 13, 14, 15);
}

__device__ __forceinline__ v16bf frag_from_bf16(const __bf16* p, const __bf16* q) {
    v8bf lo = *(const v8bf*)p;          // 16B contiguous, 16B-aligned
    v8bf hi = *(const v8bf*)q;
    return __builtin_shufflevector(lo, hi,
        0, 1, 2, 3, 4, 5, 6, 7, 8, 9, 10, 11, 12, 13, 14, 15);
}

__device__ __forceinline__ v8f wmma_bf16(v16bf a, v16bf b, v8f c) {
    return __builtin_amdgcn_wmma_f32_16x16x32_bf16(
        /*neg_a=*/false, a, /*neg_b=*/false, b,
        /*c_mod=*/(short)0, c, /*reuse_a=*/false, /*reuse_b=*/false);
}

// ---------------------------------------------------------------------------
// Kernel 0: init -- zero the grid-sync counter and the h0 state (graph-safe)
// ---------------------------------------------------------------------------
__global__ void init_kernel(unsigned* __restrict__ sync, uint32_t* __restrict__ hA_u32) {
    int i = blockIdx.x * blockDim.x + threadIdx.x;
    if (i == 0) *sync = 0u;
    if (i < (BATCH * HIDDEN) / 2) hA_u32[i] = 0u;   // 64*1024 bf16 = 32768 dwords
}

// ---------------------------------------------------------------------------
// Kernel 1: wx = gather(emb, x) @ W^T + b_w   -> bf16 [32768, 1024]
// One wave computes a 32x32 output block (2x2 WMMA tiles), K = 512.
// Grid: 4096 WGs x 256 threads (8 waves) = 32768 waves = 1024 Mgrp x 32 Ngrp.
// ---------------------------------------------------------------------------
__global__ void __launch_bounds__(256) wx_gemm_kernel(
    const int*   __restrict__ x,
    const float* __restrict__ emb,
    const float* __restrict__ W,
    const float* __restrict__ b_w,
    __bf16*      __restrict__ wx)
{
    const int lane = threadIdx.x & 31;
    const int gw   = blockIdx.x * 8 + (threadIdx.x >> 5);
    const int m0   = (gw >> 5) * 32;     // row block  (of 32768)
    const int n0   = (gw & 31) * 32;     // col block  (of 1024)

    // Tokens for this wave's two 16-row A sub-tiles (gather rows of emb)
    const int    rA0   = m0 + (lane & 15);
    const float* arow0 = emb + (size_t)x[rA0]      * EMBED;
    const float* arow1 = emb + (size_t)x[rA0 + 16] * EMBED;

    const int koffA = (lane & 16) ? 8  : 0;   // A fragment K sub-offset
    const int koffB = (lane & 16) ? 8  : 0;   // B fragment K sub-offset
    const int nB    = lane & 15;
    const float* brow0 = W + (size_t)(n0 + nB)      * EMBED;  // W rows = B cols
    const float* brow1 = W + (size_t)(n0 + 16 + nB) * EMBED;

    v8f c00 = {}, c01 = {}, c10 = {}, c11 = {};

#pragma unroll 2
    for (int k0 = 0; k0 < EMBED; k0 += 32) {
        v16bf a0 = frag_from_f32(arow0 + k0 + koffA, arow0 + k0 + koffA + 16);
        v16bf a1 = frag_from_f32(arow1 + k0 + koffA, arow1 + k0 + koffA + 16);
        v16bf b0 = frag_from_f32(brow0 + k0 + koffB, brow0 + k0 + koffB + 16);
        v16bf b1 = frag_from_f32(brow1 + k0 + koffB, brow1 + k0 + koffB + 16);
        c00 = wmma_bf16(a0, b0, c00);
        c01 = wmma_bf16(a0, b1, c01);
        c10 = wmma_bf16(a1, b0, c10);
        c11 = wmma_bf16(a1, b1, c11);
    }

    const float bw0 = b_w[n0 + nB];
    const float bw1 = b_w[n0 + 16 + nB];
    const int   mo  = m0 + ((lane & 16) ? 8 : 0);
    const int   h0c = n0 + nB;
#pragma unroll
    for (int r = 0; r < 8; ++r) {
        wx[(size_t)(mo + r)      * HIDDEN + h0c]      = (__bf16)(c00[r] + bw0);
        wx[(size_t)(mo + r)      * HIDDEN + h0c + 16] = (__bf16)(c01[r] + bw1);
        wx[(size_t)(mo + r + 16) * HIDDEN + h0c]      = (__bf16)(c10[r] + bw0);
        wx[(size_t)(mo + r + 16) * HIDDEN + h0c + 16] = (__bf16)(c11[r] + bw1);
    }
}

// ---------------------------------------------------------------------------
// Kernel 2: persistent recurrence  h = tanh(wx_t + h @ U^T), 512 steps.
// 64 WGs x 128 threads; WG owns 16 columns, U^T tile lives in LDS (bf16).
// Two independent WMMA accumulator chains per wave (even/odd k-steps) so
// dependent WMMAs never issue back-to-back on the serial critical path.
// Grid-wide barrier per step via monotonic atomic counter.
// ---------------------------------------------------------------------------
__global__ void __launch_bounds__(128) rnn_kernel(
    const __bf16* __restrict__ wx,
    const float*  __restrict__ U,
    __bf16*       __restrict__ hA,
    __bf16*       __restrict__ hB,
    unsigned*     __restrict__ sync)
{
    __shared__ __bf16 Ut[16 * ULDS_STRIDE];

    const int lane = threadIdx.x & 31;
    const int wave = threadIdx.x >> 5;
    const int n0   = blockIdx.x * 16;        // this WG's output columns
    const int m0   = wave * 16;              // this wave's batch rows

    // Stage U[n0..n0+15][0..1023] into LDS as bf16 (resident for all steps)
    for (int i = threadIdx.x; i < 16 * HIDDEN; i += 128) {
        int r = i >> 10, c = i & (HIDDEN - 1);
        Ut[r * ULDS_STRIDE + c] = (__bf16)U[(size_t)(n0 + r) * HIDDEN + c];
    }
    __syncthreads();

    const int koffA = (lane & 16) ? 8  : 0;
    const int koffB = (lane & 16) ? 8  : 0;
    const int nB    = lane & 15;
    const __bf16* Urow = &Ut[nB * ULDS_STRIDE];
    const int bRow  = m0 + (lane & 15);      // A-fragment row (batch index)
    const int hcol  = n0 + nB;               // output column
    const int mo    = m0 + ((lane & 16) ? 8 : 0);

    for (int t = 0; t < SEQ; ++t) {
        const __bf16* hc = (t & 1) ? hB : hA;
        __bf16*       hn = (t & 1) ? hA : hB;

        const __bf16* arow = hc + (size_t)bRow * HIDDEN;
        v8f acc0 = {}, acc1 = {};
#pragma unroll 4
        for (int k0 = 0; k0 < HIDDEN; k0 += 64) {     // 2 k-steps / iter
            v16bf a0 = frag_from_bf16(arow + k0 + koffA,
                                      arow + k0 + koffA + 16);
            v16bf b0 = frag_from_bf16(Urow + k0 + koffB,
                                      Urow + k0 + koffB + 16);
            v16bf a1 = frag_from_bf16(arow + k0 + 32 + koffA,
                                      arow + k0 + 32 + koffA + 16);
            v16bf b1 = frag_from_bf16(Urow + k0 + 32 + koffB,
                                      Urow + k0 + 32 + koffB + 16);
            acc0 = wmma_bf16(a0, b0, acc0);           // independent chains
            acc1 = wmma_bf16(a1, b1, acc1);
        }

#pragma unroll
        for (int r = 0; r < 8; ++r) {
            int brow = mo + r;
            float v = acc0[r] + acc1[r] +
                      (float)wx[((size_t)brow * SEQ + t) * HIDDEN + hcol];
            hn[(size_t)brow * HIDDEN + hcol] = (__bf16)tanhf(v);
        }

        // ---- grid-wide barrier (monotonic counter, graph-replay safe) ----
        __syncthreads();
        if (threadIdx.x == 0) {
            __threadfence();                       // release h writes
            atomicAdd(sync, 1u);
            const unsigned target = (unsigned)(t + 1) * RNN_WGS;
            while (__hip_atomic_load(sync, __ATOMIC_RELAXED,
                                     __HIP_MEMORY_SCOPE_AGENT) < target) {
                __builtin_amdgcn_s_sleep(2);
            }
            __threadfence();                       // acquire other WGs' writes
        }
        __syncthreads();
    }
}

// ---------------------------------------------------------------------------
// Kernel 3: head  out = h_final @ V^T + b_v   ([64,1024] x [1024,4])
// Final state is in hA (512 steps: last write at t=511 targets hA).
// ---------------------------------------------------------------------------
__global__ void __launch_bounds__(256) head_kernel(
    const __bf16* __restrict__ h,
    const float*  __restrict__ V,
    const float*  __restrict__ b_v,
    float*        __restrict__ out)
{
    const int t = threadIdx.x;          // 256 = 64 batch x 4 classes
    const int b = t >> 2, c = t & 3;
    const __bf16* hr = h + (size_t)b * HIDDEN;
    const float*  vr = V + (size_t)c * HIDDEN;
    float s = b_v[c];
#pragma unroll 4
    for (int k = 0; k < HIDDEN; ++k) s += (float)hr[k] * vr[k];
    out[b * NCLS + c] = s;
}

// ---------------------------------------------------------------------------
// Host-side launch
// ---------------------------------------------------------------------------
extern "C" void kernel_launch(void* const* d_in, const int* in_sizes, int n_in,
                              void* d_out, int out_size, void* d_ws, size_t ws_size,
                              hipStream_t stream) {
    const int*   x   = (const int*)  d_in[0];
    const float* emb = (const float*)d_in[1];
    const float* W   = (const float*)d_in[2];
    const float* b_w = (const float*)d_in[3];
    const float* U   = (const float*)d_in[4];
    const float* V   = (const float*)d_in[5];
    const float* b_v = (const float*)d_in[6];
    float*       out = (float*)d_out;

    // workspace layout (all 256B-aligned):
    //   [0,256)                        sync counter
    //   [256, 256+128K)                hA  (bf16 64x1024)
    //   [256+128K, 256+256K)           hB  (bf16 64x1024)
    //   [256+256K, 256+256K+64M)       wx  (bf16 32768x1024)
    char*     ws   = (char*)d_ws;
    unsigned* sync = (unsigned*)ws;
    __bf16*   hA   = (__bf16*)(ws + 256);
    __bf16*   hB   = (__bf16*)(ws + 256 + (size_t)BATCH * HIDDEN * 2);
    __bf16*   wx   = (__bf16*)(ws + 256 + (size_t)2 * BATCH * HIDDEN * 2);

    init_kernel<<<128, 256, 0, stream>>>(sync, (uint32_t*)hA);
    wx_gemm_kernel<<<MROWS / 32 * (HIDDEN / 32) / 8, 256, 0, stream>>>(x, emb, W, b_w, wx);
    rnn_kernel<<<RNN_WGS, 128, 0, stream>>>(wx, U, hA, hB, sync);
    head_kernel<<<1, 256, 0, stream>>>(hA, V, b_v, out);
}
